// ViT_11690900980295
// MI455X (gfx1250) — matrix operand order repaired and checked by reading, not compiled, not run on hardware
//
#include <hip/hip_runtime.h>
#include <hip/hip_bf16.h>

typedef __attribute__((ext_vector_type(16))) _Float16 v16h;
typedef __attribute__((ext_vector_type(8)))  _Float16 v8h;
typedef __attribute__((ext_vector_type(8)))  float    v8f;

#define DEV __device__ __forceinline__

// ---- model dims ----
#define BB 32
#define CC 3
#define HH 224
#define PP 14
#define PSZ 16
#define DD 768
#define NHD 12
#define DHD 64
#define LL 12
#define FFD 3072
#define OUTD 1000
#define SS 197
#define SP 208          // padded token count (13*16)
#define KP 224          // padded attention K stride (7*32)
#define NPATCH 196
#define MROWS (BB*SS)   // 6304 = 394*16
#define MTOK (BB*NPATCH) // 6272 = 392*16
#define NOPAD 1008      // padded OUT (63*16)

DEV int globalWave() { return blockIdx.x * (blockDim.x >> 5) + (threadIdx.x >> 5); }

DEV float wred_sum(float v) {
#pragma unroll
  for (int o = 16; o >= 1; o >>= 1) v += __shfl_xor(v, o, 32);
  return v;
}
DEV float wred_max(float v) {
#pragma unroll
  for (int o = 16; o >= 1; o >>= 1) v = fmaxf(v, __shfl_xor(v, o, 32));
  return v;
}

// A fragment (16x32 f16): lane = row (lane&15); lanes 0-15 hold K k0..k0+7 and
// k0+16..k0+23 with k0=0; lanes 16-31 same with k0=8.  Two 16B loads per lane.
DEV v16h ld_fragA(const _Float16* __restrict__ tile, int ld) {
  int lane = threadIdx.x & 31;
  const _Float16* p = tile + (size_t)(lane & 15) * ld + ((lane >> 4) << 3);
  v8h lo = *reinterpret_cast<const v8h*>(p);
  v8h hi = *reinterpret_cast<const v8h*>(p + 16);
  v16h a;
#pragma unroll
  for (int i = 0; i < 8; ++i) { a[i] = lo[i]; a[i + 8] = hi[i]; }
  return a;
}

// B fragment (32x16 f16), sourced from BT ([N,K] row-major): lane = col (lane&15);
// lanes 0-15 hold K=0..7 and 16..23, lanes 16-31 hold K=8..15 and 24..31 (matches A).
DEV v16h ld_fragB(const _Float16* __restrict__ tile, int ld) {
  int lane = threadIdx.x & 31;
  const _Float16* p = tile + (size_t)(lane & 15) * ld + ((lane >> 4) << 3);
  v8h lo = *reinterpret_cast<const v8h*>(p);
  v8h hi = *reinterpret_cast<const v8h*>(p + 16);
  v16h b;
#pragma unroll
  for (int i = 0; i < 8; ++i) { b[i] = lo[i]; b[i + 8] = hi[i]; }
  return b;
}

// single 16x16 tile (used by the small/ragged attention GEMMs)
DEV v8f wmma_tile(const _Float16* __restrict__ A, int lda,
                  const _Float16* __restrict__ BT, int ldb, int K) {
  v8f acc = {};
  for (int kk = 0; kk < K; kk += 32) {
    v16h a = ld_fragA(A + kk, lda);
    v16h b = ld_fragB(BT + kk, ldb);
    acc = __builtin_amdgcn_wmma_f32_16x16x32_f16(false, a, false, b,
                                                 (short)0, acc, false, false);
  }
  return acc;
}

// 2x4 register-blocked macro-tile: 32 rows x 64 cols per wave.
// 6 fragment loads feed 8 WMMAs per K-step (~22 FLOP/B of cache traffic).
DEV void wmma_tile_2x4(const _Float16* __restrict__ A, int lda,
                       const _Float16* __restrict__ BT, int ldb, int K,
                       v8f acc[2][4]) {
#pragma unroll
  for (int i = 0; i < 2; ++i)
#pragma unroll
    for (int j = 0; j < 4; ++j) acc[i][j] = (v8f){};
  for (int kk = 0; kk < K; kk += 32) {
    v16h a0 = ld_fragA(A + kk, lda);
    v16h a1 = ld_fragA(A + (size_t)16 * lda + kk, lda);
    v16h b0 = ld_fragB(BT + kk, ldb);
    v16h b1 = ld_fragB(BT + (size_t)16 * ldb + kk, ldb);
    v16h b2 = ld_fragB(BT + (size_t)32 * ldb + kk, ldb);
    v16h b3 = ld_fragB(BT + (size_t)48 * ldb + kk, ldb);
    acc[0][0] = __builtin_amdgcn_wmma_f32_16x16x32_f16(false, a0, false, b0, (short)0, acc[0][0], false, false);
    acc[0][1] = __builtin_amdgcn_wmma_f32_16x16x32_f16(false, a0, false, b1, (short)0, acc[0][1], false, false);
    acc[0][2] = __builtin_amdgcn_wmma_f32_16x16x32_f16(false, a0, false, b2, (short)0, acc[0][2], false, false);
    acc[0][3] = __builtin_amdgcn_wmma_f32_16x16x32_f16(false, a0, false, b3, (short)0, acc[0][3], false, false);
    acc[1][0] = __builtin_amdgcn_wmma_f32_16x16x32_f16(false, a1, false, b0, (short)0, acc[1][0], false, false);
    acc[1][1] = __builtin_amdgcn_wmma_f32_16x16x32_f16(false, a1, false, b1, (short)0, acc[1][1], false, false);
    acc[1][2] = __builtin_amdgcn_wmma_f32_16x16x32_f16(false, a1, false, b2, (short)0, acc[1][2], false, false);
    acc[1][3] = __builtin_amdgcn_wmma_f32_16x16x32_f16(false, a1, false, b3, (short)0, acc[1][3], false, false);
  }
}

DEV float posenc(int s, int d) {
  float je = (float)(d & ~1);
  float ang = (float)s * powf(10000.0f, -je * (1.0f / 768.0f));
  return (d & 1) ? cosf(ang) : sinf(ang);
}

// ---------------- elementwise / setup kernels ----------------

__global__ void k_convert(const float* __restrict__ src, _Float16* __restrict__ dst,
                          int n_src, int n_total) {
  int idx = blockIdx.x * blockDim.x + threadIdx.x;
  if (idx >= n_total) return;
  dst[idx] = (idx < n_src) ? (_Float16)src[idx] : (_Float16)0.0f;
}

__global__ void k_patchify(const float* __restrict__ img, _Float16* __restrict__ out) {
  int idx = blockIdx.x * blockDim.x + threadIdx.x;
  if (idx >= MTOK * DD) return;
  int row = idx / DD, cps = idx % DD;
  int b = row / NPATCH, p = row % NPATCH;
  int py = p / PP, px = p % PP;
  int c = cps >> 8, rr = (cps >> 4) & 15, wc = cps & 15;
  size_t src = (((size_t)(b * CC + c) * HH) + py * PSZ + rr) * HH + px * PSZ + wc;
  out[idx] = (_Float16)img[src];
}

__global__ void k_cls_pos(const float* __restrict__ cls, float* __restrict__ x) {
  int idx = blockIdx.x * blockDim.x + threadIdx.x;
  if (idx >= BB * DD) return;
  int b = idx / DD, d = idx % DD;
  x[(size_t)b * SS * DD + d] = cls[d] + posenc(0, d);
}

__global__ void k_extract_cls(const float* __restrict__ x, _Float16* __restrict__ xc) {
  int idx = blockIdx.x * blockDim.x + threadIdx.x;
  if (idx >= BB * DD) return;
  int b = idx / DD, d = idx % DD;
  xc[idx] = (_Float16)x[(size_t)b * SS * DD + d];
}

// ---------------- row-wise kernels ----------------

__global__ __launch_bounds__(256) void k_layernorm(const float* __restrict__ x,
    const float* __restrict__ g, const float* __restrict__ bta,
    _Float16* __restrict__ out, int rows) {
  int row = globalWave();
  if (row >= rows) return;
  int lane = threadIdx.x & 31;
  const float* xr = x + (size_t)row * DD;
  float s = 0.f, s2 = 0.f;
  for (int i = lane; i < DD; i += 32) { float v = xr[i]; s += v; s2 += v * v; }
  s = wred_sum(s); s2 = wred_sum(s2);
  float mu = s * (1.0f / DD);
  float var = s2 * (1.0f / DD) - mu * mu;
  float rs = rsqrtf(var + 1e-5f);
  _Float16* orow = out + (size_t)row * DD;
  for (int i = lane; i < DD; i += 32)
    orow[i] = (_Float16)((xr[i] - mu) * rs * g[i] + bta[i]);
}

__global__ __launch_bounds__(256) void k_softmax_attn(const float* __restrict__ sc,
                                                      _Float16* __restrict__ at) {
  int row = globalWave();
  if (row >= BB * NHD * SP) return;
  int lane = threadIdx.x & 31;
  const float* sr = sc + (size_t)row * SP;
  _Float16* ar = at + (size_t)row * KP;
  float v[7];
#pragma unroll
  for (int i = 0; i < 7; ++i) {
    int t = lane + 32 * i;
    v[i] = (t < SS) ? sr[t] : -3.0e38f;
  }
  float m = v[0];
#pragma unroll
  for (int i = 1; i < 7; ++i) m = fmaxf(m, v[i]);
  m = wred_max(m);
  float sum = 0.f;
#pragma unroll
  for (int i = 0; i < 7; ++i) {
    v[i] = (lane + 32 * i < SS) ? __expf(v[i] - m) : 0.f;
    sum += v[i];
  }
  sum = wred_sum(sum);
  float inv = 1.0f / sum;
#pragma unroll
  for (int i = 0; i < 7; ++i) {
    int t = lane + 32 * i;
    if (t < SS) ar[t] = (_Float16)(v[i] * inv);
  }
  for (int t = SS + lane; t < KP; t += 32) ar[t] = (_Float16)0.0f;  // zero pad cols
}

__global__ __launch_bounds__(256) void k_softmax_out(const float* __restrict__ lg,
                                                     float* __restrict__ out) {
  __shared__ float red[256];
  int row = blockIdx.x, tid = threadIdx.x;
  const float* lr = lg + (size_t)row * OUTD;
  float m = -3.0e38f;
  for (int i = tid; i < OUTD; i += 256) m = fmaxf(m, lr[i]);
  red[tid] = m; __syncthreads();
  for (int s = 128; s > 0; s >>= 1) { if (tid < s) red[tid] = fmaxf(red[tid], red[tid + s]); __syncthreads(); }
  m = red[0]; __syncthreads();
  float sum = 0.f;
  for (int i = tid; i < OUTD; i += 256) sum += __expf(lr[i] - m);
  red[tid] = sum; __syncthreads();
  for (int s = 128; s > 0; s >>= 1) { if (tid < s) red[tid] += red[tid + s]; __syncthreads(); }
  float inv = 1.0f / red[0];
  for (int i = tid; i < OUTD; i += 256) out[(size_t)row * OUTD + i] = __expf(lr[i] - m) * inv;
}

// ---------------- WMMA GEMM kernels ----------------

// patch embedding: [6272,768] x [768,768]^T, 2x4-blocked
__global__ __launch_bounds__(256) void k_embed(const _Float16* __restrict__ patches,
    const _Float16* __restrict__ wp, const float* __restrict__ bp, float* __restrict__ x) {
  const int NT4 = DD / 64, MT2 = MTOK / 32;  // 12 x 196
  int t = globalWave();
  if (t >= MT2 * NT4) return;
  int mt2 = t / NT4, nt4 = t % NT4;
  v8f acc[2][4];
  wmma_tile_2x4(patches + (size_t)mt2 * 32 * DD, DD, wp + (size_t)nt4 * 64 * DD, DD, DD, acc);
  int lane = threadIdx.x & 31;
  int mb = (lane >> 4) << 3;
#pragma unroll
  for (int i = 0; i < 2; ++i)
#pragma unroll
    for (int j = 0; j < 4; ++j) {
      int nn = nt4 * 64 + j * 16 + (lane & 15);
#pragma unroll
      for (int r = 0; r < 8; ++r) {
        int m = mt2 * 32 + i * 16 + mb + r;
        int b = m / NPATCH, p = m % NPATCH, s = p + 1;
        x[((size_t)(b * SS + s)) * DD + nn] = acc[i][j][r] + bp[nn] + posenc(s, nn);
      }
    }
}

__global__ __launch_bounds__(256) void k_qkv(const _Float16* __restrict__ hb,
    const _Float16* __restrict__ wq, const _Float16* __restrict__ wk,
    const _Float16* __restrict__ wv, const float* __restrict__ bq,
    const float* __restrict__ bk, const float* __restrict__ bv,
    _Float16* __restrict__ qb, _Float16* __restrict__ kb, _Float16* __restrict__ vT) {
  const int TOT = BB * NHD * 3 * 13 * 4;
  int t = globalWave();
  if (t >= TOT) return;
  int nt = t & 3; t >>= 2;
  int mt = t % 13; t /= 13;
  int sel = t % 3; t /= 3;
  int head = t % NHD, b = t / NHD;
  const _Float16* w = (sel == 0) ? wq : (sel == 1) ? wk : wv;
  const float* bias = (sel == 0) ? bq : (sel == 1) ? bk : bv;
  const _Float16* A = hb + ((size_t)(b * SS) + mt * 16) * DD + head * DHD;
  const _Float16* Bm = w + (size_t)head * DHD * DHD + nt * 16 * DHD;
  v8f acc = wmma_tile(A, DD, Bm, DHD, DHD);
  int lane = threadIdx.x & 31;
  int nn = nt * 16 + (lane & 15);
  int mb = (lane >> 4) << 3;
  int bh = b * NHD + head;
#pragma unroll
  for (int j = 0; j < 8; ++j) {
    int s = mt * 16 + mb + j;
    float val = acc[j] + bias[head * DHD + nn];
    if (sel == 2) {
      vT[((size_t)bh * DHD + nn) * KP + s] = (_Float16)val;       // store V transposed
    } else {
      _Float16* dst = sel ? kb : qb;
      dst[((size_t)bh * SP + s) * DHD + nn] = (_Float16)val;
    }
  }
}

__global__ __launch_bounds__(256) void k_scores(const _Float16* __restrict__ qb,
    const _Float16* __restrict__ kb, float* __restrict__ sc) {
  const int TOT = BB * NHD * 13 * 13;
  int t = globalWave();
  if (t >= TOT) return;
  int nt = t % 13; t /= 13;
  int mt = t % 13; int bh = t / 13;
  const _Float16* A = qb + ((size_t)bh * SP + mt * 16) * DHD;
  const _Float16* Bm = kb + ((size_t)bh * SP + nt * 16) * DHD;
  v8f acc = wmma_tile(A, DHD, Bm, DHD, DHD);
  int lane = threadIdx.x & 31;
  int nn = nt * 16 + (lane & 15);
  int mb = (lane >> 4) << 3;
  float* srow = sc + (size_t)bh * SP * SP;
#pragma unroll
  for (int j = 0; j < 8; ++j)
    srow[(size_t)(mt * 16 + mb + j) * SP + nn] = acc[j] * 0.125f;  // 1/sqrt(64)
}

__global__ __launch_bounds__(256) void k_attnv(const _Float16* __restrict__ at,
    const _Float16* __restrict__ vT, float* __restrict__ x) {
  const int TOT = BB * NHD * 13 * 4;
  int t = globalWave();
  if (t >= TOT) return;
  int nt = t & 3; t >>= 2;
  int mt = t % 13; int bh = t / 13;
  int b = bh / NHD, head = bh % NHD;
  const _Float16* A = at + ((size_t)bh * SP + mt * 16) * KP;
  const _Float16* Bm = vT + ((size_t)bh * DHD + nt * 16) * KP;
  v8f acc = wmma_tile(A, KP, Bm, KP, KP);
  int lane = threadIdx.x & 31;
  int nn = nt * 16 + (lane & 15);
  int mb = (lane >> 4) << 3;
#pragma unroll
  for (int j = 0; j < 8; ++j) {
    int s = mt * 16 + mb + j;
    if (s < SS) x[((size_t)(b * SS + s)) * DD + head * DHD + nn] += acc[j];
  }
}

// FFN1: [6304,768] x [3072,768]^T + exact GELU, 2x4-blocked
__global__ __launch_bounds__(256) void k_ffn1(const _Float16* __restrict__ hb,
    const _Float16* __restrict__ w1, const float* __restrict__ b1,
    _Float16* __restrict__ ff) {
  const int NT4 = FFD / 64, MT2 = MROWS / 32;  // 48 x 197
  int t = globalWave();
  if (t >= MT2 * NT4) return;
  int mt2 = t / NT4, nt4 = t % NT4;
  v8f acc[2][4];
  wmma_tile_2x4(hb + (size_t)mt2 * 32 * DD, DD, w1 + (size_t)nt4 * 64 * DD, DD, DD, acc);
  int lane = threadIdx.x & 31;
  int mb = (lane >> 4) << 3;
#pragma unroll
  for (int i = 0; i < 2; ++i)
#pragma unroll
    for (int j = 0; j < 4; ++j) {
      int nn = nt4 * 64 + j * 16 + (lane & 15);
#pragma unroll
      for (int r = 0; r < 8; ++r) {
        float v = acc[i][j][r] + b1[nn];
        v = 0.5f * v * (1.0f + erff(v * 0.70710678f));   // exact GELU
        ff[((size_t)(mt2 * 32 + i * 16 + mb + r)) * FFD + nn] = (_Float16)v;
      }
    }
}

// FFN2: [6304,3072] x [768,3072]^T accumulated into residual x, 2x4-blocked
__global__ __launch_bounds__(256) void k_ffn2(const _Float16* __restrict__ ff,
    const _Float16* __restrict__ w2, const float* __restrict__ b2,
    float* __restrict__ x) {
  const int NT4 = DD / 64, MT2 = MROWS / 32;  // 12 x 197
  int t = globalWave();
  if (t >= MT2 * NT4) return;
  int mt2 = t / NT4, nt4 = t % NT4;
  v8f acc[2][4];
  wmma_tile_2x4(ff + (size_t)mt2 * 32 * FFD, FFD, w2 + (size_t)nt4 * 64 * FFD, FFD, FFD, acc);
  int lane = threadIdx.x & 31;
  int mb = (lane >> 4) << 3;
#pragma unroll
  for (int i = 0; i < 2; ++i)
#pragma unroll
    for (int j = 0; j < 4; ++j) {
      int nn = nt4 * 64 + j * 16 + (lane & 15);
#pragma unroll
      for (int r = 0; r < 8; ++r) {
        size_t m = (size_t)(mt2 * 32 + i * 16 + mb + r);
        x[m * DD + nn] += acc[i][j][r] + b2[nn];
      }
    }
}

__global__ __launch_bounds__(256) void k_logits(const _Float16* __restrict__ xc,
    const _Float16* __restrict__ wo, const float* __restrict__ bo,
    float* __restrict__ lg) {
  const int NT = NOPAD / 16, MT = BB / 16;  // 63 x 2
  int t = globalWave();
  if (t >= MT * NT) return;
  int mt = t / NT, nt = t % NT;
  v8f acc = wmma_tile(xc + (size_t)mt * 16 * DD, DD, wo + (size_t)nt * 16 * DD, DD, DD);
  int lane = threadIdx.x & 31;
  int nn = nt * 16 + (lane & 15);
  int mb = (lane >> 4) << 3;
  if (nn < OUTD) {
#pragma unroll
    for (int j = 0; j < 8; ++j)
      lg[(size_t)(mt * 16 + mb + j) * OUTD + nn] = acc[j] + bo[nn];
  }
}

// ---------------- host launch ----------------

extern "C" void kernel_launch(void* const* d_in, const int* in_sizes, int n_in,
                              void* d_out, int out_size, void* d_ws, size_t ws_size,
                              hipStream_t stream) {
  (void)in_sizes; (void)n_in; (void)out_size; (void)ws_size;
  const float* images      = (const float*)d_in[0];
  const float* W_patch     = (const float*)d_in[1];
  const float* b_patch     = (const float*)d_in[2];
  const float* class_token = (const float*)d_in[3];
  const float* ln1_g       = (const float*)d_in[4];
  const float* ln1_b       = (const float*)d_in[5];
  const float* Wq          = (const float*)d_in[6];
  const float* bq          = (const float*)d_in[7];
  const float* Wk          = (const float*)d_in[8];
  const float* bk          = (const float*)d_in[9];
  const float* Wv          = (const float*)d_in[10];
  const float* bv          = (const float*)d_in[11];
  const float* ln2_g       = (const float*)d_in[12];
  const float* ln2_b       = (const float*)d_in[13];
  const float* W1          = (const float*)d_in[14];
  const float* b1          = (const float*)d_in[15];
  const float* W2          = (const float*)d_in[16];
  const float* b2          = (const float*)d_in[17];
  const float* Wo          = (const float*)d_in[18];
  const float* bo          = (const float*)d_in[19];
  float* out = (float*)d_out;

  // element counts
  const int    N_WPATCH = DD * DD;
  const int    N_WQKV   = LL * NHD * DHD * DHD;
  const int    N_W1     = LL * FFD * DD;
  const int    N_WO_SRC = OUTD * DD;
  const int    N_WO     = NOPAD * DD;
  const size_t N_X      = (size_t)MROWS * DD;
  const size_t N_H      = (size_t)(MROWS + 16) * DD;     // padded rows for qkv A-tiles
  const size_t N_QK     = (size_t)BB * NHD * SP * DHD;
  const size_t N_VT     = (size_t)BB * NHD * DHD * KP;
  const size_t N_SC     = (size_t)BB * NHD * SP * SP;    // f32; aliased with ff (f16)
  const size_t N_AT     = (size_t)BB * NHD * SP * KP;    // f16; aliased with patches

  char* ws = (char*)d_ws;
  size_t off = 0;
  auto alloc = [&](size_t bytes) -> char* {
    char* p = ws + off;
    off = (off + bytes + 255) & ~(size_t)255;
    return p;
  };

  _Float16* wp_h  = (_Float16*)alloc((size_t)N_WPATCH * 2);
  _Float16* wq_h  = (_Float16*)alloc((size_t)N_WQKV * 2);
  _Float16* wk_h  = (_Float16*)alloc((size_t)N_WQKV * 2);
  _Float16* wv_h  = (_Float16*)alloc((size_t)N_WQKV * 2);
  _Float16* w1_h  = (_Float16*)alloc((size_t)N_W1 * 2);
  _Float16* w2_h  = (_Float16*)alloc((size_t)N_W1 * 2);
  _Float16* wo_h  = (_Float16*)alloc((size_t)N_WO * 2);
  float*    x     = (float*)   alloc(N_X * 4);
  _Float16* hbuf  = (_Float16*)alloc(N_H * 2);
  _Float16* qb    = (_Float16*)alloc(N_QK * 2);
  _Float16* kb    = (_Float16*)alloc(N_QK * 2);
  _Float16* vT    = (_Float16*)alloc(N_VT * 2);
  float*    sc    = (float*)   alloc(N_SC * 4);
  _Float16* ff    = (_Float16*)sc;                 // alias: ff used after sc is dead
  _Float16* attn  = (_Float16*)alloc(N_AT * 2);
  _Float16* patches = (_Float16*)attn;             // alias: patches dead before attn
  _Float16* xcls  = (_Float16*)alloc((size_t)BB * DD * 2);
  float*    lg    = (float*)   alloc((size_t)BB * OUTD * 4);

  auto cdiv = [](int a, int b) { return (a + b - 1) / b; };

  // fp32 -> f16 weight conversion (recomputed each call; deterministic)
  k_convert<<<cdiv(N_WPATCH, 256), 256, 0, stream>>>(W_patch, wp_h, N_WPATCH, N_WPATCH);
  k_convert<<<cdiv(N_WQKV, 256), 256, 0, stream>>>(Wq, wq_h, N_WQKV, N_WQKV);
  k_convert<<<cdiv(N_WQKV, 256), 256, 0, stream>>>(Wk, wk_h, N_WQKV, N_WQKV);
  k_convert<<<cdiv(N_WQKV, 256), 256, 0, stream>>>(Wv, wv_h, N_WQKV, N_WQKV);
  k_convert<<<cdiv(N_W1, 256), 256, 0, stream>>>(W1, w1_h, N_W1, N_W1);
  k_convert<<<cdiv(N_W1, 256), 256, 0, stream>>>(W2, w2_h, N_W1, N_W1);
  k_convert<<<cdiv(N_WO, 256), 256, 0, stream>>>(Wo, wo_h, N_WO_SRC, N_WO); // zero-pads rows 1000..1007

  // patch extraction + embedding + cls/pos
  k_patchify<<<cdiv(MTOK * DD, 256), 256, 0, stream>>>(images, patches);
  k_cls_pos<<<cdiv(BB * DD, 256), 256, 0, stream>>>(class_token, x);
  {
    int nw = (MTOK / 32) * (DD / 64);
    k_embed<<<cdiv(nw, 8), 256, 0, stream>>>(patches, wp_h, b_patch, x);
  }

  for (int l = 0; l < LL; ++l) {
    k_layernorm<<<cdiv(MROWS, 8), 256, 0, stream>>>(x, ln1_g + l * DD, ln1_b + l * DD, hbuf, MROWS);
    {
      int nw = BB * NHD * 3 * 13 * 4;
      k_qkv<<<cdiv(nw, 8), 256, 0, stream>>>(hbuf,
          wq_h + (size_t)l * NHD * DHD * DHD,
          wk_h + (size_t)l * NHD * DHD * DHD,
          wv_h + (size_t)l * NHD * DHD * DHD,
          bq + (size_t)l * NHD * DHD, bk + (size_t)l * NHD * DHD, bv + (size_t)l * NHD * DHD,
          qb, kb, vT);
    }
    {
      int nw = BB * NHD * 13 * 13;
      k_scores<<<cdiv(nw, 8), 256, 0, stream>>>(qb, kb, sc);
    }
    {
      int nw = BB * NHD * SP;
      k_softmax_attn<<<cdiv(nw, 8), 256, 0, stream>>>(sc, attn);
    }
    {
      int nw = BB * NHD * 13 * 4;
      k_attnv<<<cdiv(nw, 8), 256, 0, stream>>>(attn, vT, x);
    }
    k_layernorm<<<cdiv(MROWS, 8), 256, 0, stream>>>(x, ln2_g + l * DD, ln2_b + l * DD, hbuf, MROWS);
    {
      int nw = (MROWS / 32) * (FFD / 64);
      k_ffn1<<<cdiv(nw, 8), 256, 0, stream>>>(hbuf, w1_h + (size_t)l * FFD * DD, b1 + (size_t)l * FFD, ff);
    }
    {
      int nw = (MROWS / 32) * (DD / 64);
      k_ffn2<<<cdiv(nw, 8), 256, 0, stream>>>(ff, w2_h + (size_t)l * DD * FFD, b2 + (size_t)l * DD, x);
    }
  }

  k_extract_cls<<<cdiv(BB * DD, 256), 256, 0, stream>>>(x, xcls);
  {
    int nw = (BB / 16) * (NOPAD / 16);
    k_logits<<<cdiv(nw, 8), 256, 0, stream>>>(xcls, wo_h, bo, lg);
  }
  k_softmax_out<<<BB, 256, 0, stream>>>(lg, out);
}
